// ELM_AE_Spatial_EVOLVED_36490042147429
// MI455X (gfx1250) — matrix operand order repaired and checked by reading, not compiled, not run on hardware
//
#include <hip/hip_runtime.h>
#include <math.h>

typedef __attribute__((ext_vector_type(16))) _Float16 v16h;
typedef __attribute__((ext_vector_type(8)))  float    v8f;

#define BB      32
#define CC      2048
#define NPATCH  36            // n' per (b',c')
#define NGROUP  (BB * NPATCH) // 1152 groups (b',n')
#define NPAIR   (BB * CC)     // 65536 (b',c') pairs

__device__ __forceinline__ int clampi(int v, int lo, int hi) {
    return v < lo ? lo : (v > hi ? hi : v);
}

// Patch value at resized position (row=2*nh+ki, col=2*nw+kj), computed
// directly from the 7x7 source tile (bilinear, align_corners=False, clamped).
__device__ __forceinline__ float bilin4(const float* __restrict__ tile,
                                        int nh, int nw, int ki, int kj) {
    int li0 = (ki + 1) >> 1, lj0 = (kj + 1) >> 1;
    int gy0 = clampi(nh - 1 + li0, 0, 6);
    int gy1 = clampi(nh     + li0, 0, 6);
    int gx0 = clampi(nw - 1 + lj0, 0, 6);
    int gx1 = clampi(nw     + lj0, 0, 6);
    float wy = (ki & 1) ? 0.25f : 0.75f;
    float wx = (kj & 1) ? 0.25f : 0.75f;
    float a00 = tile[gy0 * 7 + gx0], a01 = tile[gy0 * 7 + gx1];
    float a10 = tile[gy1 * 7 + gx0], a11 = tile[gy1 * 7 + gx1];
    float top = a00 + wx * (a01 - a00);
    float bot = a10 + wx * (a11 - a10);
    return top + wy * (bot - top);
}

// ---------------------------------------------------------------------------
// Kernel 1: per-group (b',n') mean / inv-std over the 2048 scrambled channels,
// for each of the 16 patch positions. One 256-thread block per group.
// ---------------------------------------------------------------------------
__global__ void radam_stats(const float* __restrict__ x_last,
                            float2* __restrict__ stats) {   // [1152][16]
    const int group = blockIdx.x;            // b'*36 + n'
    const int bP = group / NPATCH;
    const int nP = group - bP * NPATCH;
    const int tid = threadIdx.x;             // 256

    float sum[16], ssq[16];
#pragma unroll
    for (int p = 0; p < 16; ++p) { sum[p] = 0.f; ssq[p] = 0.f; }

    for (int cP = tid; cP < CC; cP += 256) {
        int g = (bP * CC + cP) * NPATCH + nP;
        int nh = g / 393216;  int r  = g - nh * 393216;   // 393216 = 6*32*2048
        int nw = r >> 16;     int r2 = r & 65535;
        int b  = r2 >> 11;    int c  = r2 & 2047;
        const float* tile = x_last + (((size_t)b << 11) + (size_t)c) * 49;

        // 4x4 clamped source window around (nh, nw)
        float vwin[4][4];
#pragma unroll
        for (int i = 0; i < 4; ++i) {
            int gy = clampi(nh - 1 + i, 0, 6);
#pragma unroll
            for (int j = 0; j < 4; ++j) {
                int gx = clampi(nw - 1 + j, 0, 6);
                vwin[i][j] = tile[gy * 7 + gx];
            }
        }
#pragma unroll
        for (int ki = 0; ki < 4; ++ki) {
            int li0 = (ki + 1) >> 1;
            float wy = (ki & 1) ? 0.25f : 0.75f;
#pragma unroll
            for (int kj = 0; kj < 4; ++kj) {
                int lj0 = (kj + 1) >> 1;
                float wx = (kj & 1) ? 0.25f : 0.75f;
                float top = vwin[li0][lj0]     + wx * (vwin[li0][lj0 + 1]     - vwin[li0][lj0]);
                float bot = vwin[li0 + 1][lj0] + wx * (vwin[li0 + 1][lj0 + 1] - vwin[li0 + 1][lj0]);
                float v = top + wy * (bot - top);
                int p = ki * 4 + kj;
                sum[p] += v; ssq[p] += v * v;
            }
        }
    }

    // wave32 reduction of the 32 accumulators
#pragma unroll
    for (int off = 16; off > 0; off >>= 1) {
#pragma unroll
        for (int p = 0; p < 16; ++p) {
            sum[p] += __shfl_xor(sum[p], off, 32);
            ssq[p] += __shfl_xor(ssq[p], off, 32);
        }
    }
    __shared__ float lsum[8][16];
    __shared__ float lssq[8][16];
    const int wave = tid >> 5, lane = tid & 31;
    if (lane == 0) {
#pragma unroll
        for (int p = 0; p < 16; ++p) { lsum[wave][p] = sum[p]; lssq[wave][p] = ssq[p]; }
    }
    __syncthreads();
    if (tid < 16) {
        float S = 0.f, Q = 0.f;
#pragma unroll
        for (int w = 0; w < 8; ++w) { S += lsum[w][tid]; Q += lssq[w][tid]; }
        float mean = S * (1.0f / CC);
        float var  = Q * (1.0f / CC) - mean * mean;
        float sd   = sqrtf(fmaxf(var, 0.f));
        float inv  = (sd > 0.f) ? (1.0f / sd) : 0.f;  // nan_to_num(0/0) -> 0
        stats[group * 16 + tid] = make_float2(mean, inv);
    }
}

// ---------------------------------------------------------------------------
// Kernel 2: one wave32 per (b',c') pair. Build Xs(16x36) in LDS, zscore over
// the 16 features, tanh hidden row, then num = Xz * Zh via two
// V_WMMA_F32_16X16X32_F16 (Zh broadcast across the 16 B-columns).
// ---------------------------------------------------------------------------
__global__ void radam_elm(const float*  __restrict__ x_last,
                          const float*  __restrict__ W_elm,   // 17
                          const float2* __restrict__ stats,   // [1152][16]
                          float*        __restrict__ out) {   // 65536
    __shared__ float sXs[8][16][37];   // [wave][p][n]  (padded)
    __shared__ float sZh[8][40];       // [wave][n]     (36 + zero pad)
    __shared__ float sW[17];

    const int tid  = threadIdx.x;
    const int wave = tid >> 5;
    const int lane = tid & 31;
    if (tid < 17) sW[tid] = W_elm[tid];
    if (lane < 8) sZh[wave][32 + lane] = 0.f;

    const int pid = blockIdx.x * 8 + wave;   // b'*2048 + c'
    const int bP  = pid >> 11;

    // Phase 1: fill Xs (channel-zscored patch values). 576 entries / wave.
#pragma unroll 2
    for (int it = 0; it < 18; ++it) {
        int idx = lane + 32 * it;            // 0..575
        int n = idx >> 4, p = idx & 15;
        int g = pid * NPATCH + n;
        int nh = g / 393216;  int r  = g - nh * 393216;
        int nw = r >> 16;     int r2 = r & 65535;
        int b  = r2 >> 11;    int c  = r2 & 2047;
        const float* tile = x_last + (((size_t)b << 11) + (size_t)c) * 49;
        float v = bilin4(tile, nh, nw, p >> 2, p & 3);
        float2 st = stats[(bP * NPATCH + n) * 16 + p];
        sXs[wave][p][n] = (v - st.x) * st.y;
    }
    __syncthreads();

    // Phase 2: per-column (n) zscore over 16 features + hidden Zh = tanh(W.Xb)
    for (int n = lane; n < NPATCH; n += 32) {
        float m = 0.f;
#pragma unroll
        for (int p = 0; p < 16; ++p) m += sXs[wave][p][n];
        m *= (1.0f / 16.0f);
        float q = 0.f;
#pragma unroll
        for (int p = 0; p < 16; ++p) { float d = sXs[wave][p][n] - m; q += d * d; }
        float sd  = sqrtf(q * (1.0f / 16.0f));
        float inv = (sd > 0.f) ? (1.0f / sd) : 0.f;
        float acc = sW[0];                 // bias row of W_elm
#pragma unroll
        for (int p = 0; p < 16; ++p) {
            float z = (sXs[wave][p][n] - m) * inv;
            sXs[wave][p][n] = z;
            acc += sW[p + 1] * z;
        }
        sZh[wave][n] = tanhf(acc);
    }
    __syncthreads();

    // den = ||Zh||^2 + lambda  (wave reduction)
    float dd = sZh[wave][lane] * sZh[wave][lane];
    if (lane < 4) { float z = sZh[wave][32 + lane]; dd += z * z; }
#pragma unroll
    for (int off = 16; off > 0; off >>= 1) dd += __shfl_xor(dd, off, 32);
    const float den = dd + 0.001f;

    // Build WMMA fragments. A = Xz (16x36, K padded to 64). B = Zh broadcast
    // over 16 columns. Layouts per CDNA5 ISA 7.12.2 (wave32).
    const bool hi = (lane >= 16);
    const int  M  = lane & 15;
    v16h a0, a1, b0, b1;
#pragma unroll
    for (int j = 0; j < 8; ++j) {
        int k0 = (hi ? 8 : 0) + j;          // K 0..15
        int k1 = (hi ? 24 : 16) + j;        // K 16..31
        a0[j]     = (_Float16)sXs[wave][M][k0];
        a0[8 + j] = (_Float16)sXs[wave][M][k1];
        int k2 = 32 + k0;                   // K 32..47
        a1[j]     = (k2 < NPATCH) ? (_Float16)sXs[wave][M][k2] : (_Float16)0.f;
        a1[8 + j] = (_Float16)0.f;          // K 48..63 all zero
    }
#pragma unroll
    for (int t = 0; t < 16; ++t) {
        int k = (t < 8) ? ((hi ? 8 : 0) + t) : ((hi ? 24 : 16) + (t - 8)); // K 0..31 (table order)
        b0[t] = (_Float16)sZh[wave][k];
        int k2 = 32 + k;                    // K 32..63
        b1[t] = (k2 < 40) ? (_Float16)sZh[wave][k2] : (_Float16)0.f; // 36..39 = 0 pad
    }

    v8f cacc = {};
    cacc = __builtin_amdgcn_wmma_f32_16x16x32_f16(false, a0, false, b0,
                                                  (short)0, cacc, false, false);
    cacc = __builtin_amdgcn_wmma_f32_16x16x32_f16(false, a1, false, b1,
                                                  (short)0, cacc, false, false);

    // D[p][j] = num[p] for every column j. Lanes 0-15 hold M=0..7 (per VGPR r),
    // lanes 16-31 hold M=8..15. Sum registers, then pair with lane^16.
    float s = 0.f;
#pragma unroll
    for (int r = 0; r < 8; ++r) s += cacc[r];
    s += __shfl_xor(s, 16, 32);

    if (lane == 0) out[pid] = s / (16.0f * den);
}

// ---------------------------------------------------------------------------
extern "C" void kernel_launch(void* const* d_in, const int* in_sizes, int n_in,
                              void* d_out, int out_size, void* d_ws, size_t ws_size,
                              hipStream_t stream) {
    (void)in_sizes; (void)n_in; (void)out_size; (void)ws_size;
    // d_in[0] = x_mid (unused by the math; only its spatial size matters)
    const float* x_last = (const float*)d_in[1];
    const float* W_elm  = (const float*)d_in[2];
    float2* stats = (float2*)d_ws;            // 1152*16 float2 = 147,456 B
    float*  out   = (float*)d_out;            // 65536 fp32

    radam_stats<<<NGROUP, 256, 0, stream>>>(x_last, stats);
    radam_elm<<<NPAIR / 8, 256, 0, stream>>>(x_last, W_elm, stats, out);
}
